// SymplecticLeapfrogLayer_6597069767469
// MI455X (gfx1250) — matrix operand order repaired and checked by reading, not compiled, not run on hardware
//
#include <hip/hip_runtime.h>
#include <hip/hip_bf16.h>

typedef __attribute__((ext_vector_type(16))) __bf16 v16bf;
typedef __attribute__((ext_vector_type(8)))  __bf16 v8bf;
typedef __attribute__((ext_vector_type(8)))  float  v8f;

#define BN_EPS_F 1e-3f
#define DT_F     0.1f
#define HALF_DT  0.05f

union V16U { v16bf v; v8bf h[2]; };

#define WMMA_BF16(a, b, c) \
    __builtin_amdgcn_wmma_f32_16x16x32_bf16(false, (a), false, (b), (short)0, (c), false, false)

// Branch-free tanh on raw TRANS units. Saturates naturally at +-1.
__device__ __forceinline__ float tanh_fast(float x) {
#if __has_builtin(__builtin_amdgcn_tanhf)
    return __builtin_amdgcn_tanhf(x);        // CDNA5 native v_tanh_f32
#else
    // tanh(x) = 1 - 2/(exp(2x)+1);  v_exp_f32 is exp2 -> scale by 2*log2(e)
    const float e = __builtin_amdgcn_exp2f(x * 2.8853900817779268f);
    return 1.0f - 2.0f * __builtin_amdgcn_rcpf(e + 1.0f);
#endif
}

__device__ __forceinline__ float rsq_fast(float x) { return __builtin_amdgcn_rsqf(x); }

// A-tile (16x32 bf16) from LDS, row-major, row stride K halfs, rows m0..m0+15.
// ISA layout: lane m(<16): K 0..7 & 16..23 ; lane m+16: K 8..15 & 24..31.
template<int K>
__device__ __forceinline__ v16bf load_A(const __bf16* A, int m0, int k0, int lane) {
    const int m  = m0 + (lane & 15);
    const int kh = (lane >> 4) << 3;
    const __bf16* p = A + m * K + k0 + kh;
    V16U u;
    u.h[0] = *(const v8bf*)(p);
    u.h[1] = *(const v8bf*)(p + 16);
    return u.v;
}

// B-tile (32x16 bf16) from global, column-major, column stride K halfs.
// ISA layout: lane n(<16): col n rows k0..k0+7 & k0+16..k0+23 ; lane n+16: rows k0+8..k0+15 & k0+24..k0+31.
template<int K>
__device__ __forceinline__ v16bf load_B(const __bf16* B, int bc, int k0, int lane) {
    const int c  = lane & 15;
    const int kr = (lane >> 4) << 3;
    const __bf16* p = B + (bc + c) * K + k0 + kr;
    V16U u;
    u.h[0] = *(const v8bf*)(p);
    u.h[1] = *(const v8bf*)(p + 16);
    return u.v;
}

// C[m-tile0], C[m-tile1] += A(32xK) * B tile: one B load feeds two WMMAs.
template<int K>
__device__ __forceinline__ void mm2(const __bf16* Alds, const __bf16* Bg, int bc, int lane,
                                    v8f& c0, v8f& c1) {
#pragma unroll 4
    for (int k0 = 0; k0 < K; k0 += 32) {
        v16bf b  = load_B<K>(Bg, bc, k0, lane);
        v16bf a0 = load_A<K>(Alds, 0,  k0, lane);
        v16bf a1 = load_A<K>(Alds, 16, k0, lane);
        c0 = WMMA_BF16(a0, b, c0);
        c1 = WMMA_BF16(a1, b, c1);
    }
}

// Two B matrices, shared A loads, both M-tiles: 4 WMMAs per (2 ds + 2 global) b256 loads.
template<int K>
__device__ __forceinline__ void mm2x2(const __bf16* Alds, const __bf16* B0, const __bf16* B1,
                                      int bc, int lane,
                                      v8f& c00, v8f& c01, v8f& c10, v8f& c11) {
#pragma unroll 2
    for (int k0 = 0; k0 < K; k0 += 32) {
        v16bf b0 = load_B<K>(B0, bc, k0, lane);
        v16bf b1 = load_B<K>(B1, bc, k0, lane);
        v16bf a0 = load_A<K>(Alds, 0,  k0, lane);
        v16bf a1 = load_A<K>(Alds, 16, k0, lane);
        c00 = WMMA_BF16(a0, b0, c00);
        c10 = WMMA_BF16(a1, b0, c10);
        c01 = WMMA_BF16(a0, b1, c01);
        c11 = WMMA_BF16(a1, b1, c11);
    }
}

// c[m] += A0(32xK)@B0 + A1(32xK)@B1, shared B-tile streams across both M-tiles.
template<int K>
__device__ __forceinline__ void mmfin(const __bf16* A0, const __bf16* A1,
                                      const __bf16* B0, const __bf16* B1,
                                      int bc, int lane, v8f& c0, v8f& c1) {
#pragma unroll 2
    for (int k0 = 0; k0 < K; k0 += 32) {
        v16bf vb0 = load_B<K>(B0, bc, k0, lane);
        v16bf vb1 = load_B<K>(B1, bc, k0, lane);
        v16bf a00 = load_A<K>(A0, 0,  k0, lane);
        v16bf a01 = load_A<K>(A0, 16, k0, lane);
        v16bf a10 = load_A<K>(A1, 0,  k0, lane);
        v16bf a11 = load_A<K>(A1, 16, k0, lane);
        c0 = WMMA_BF16(a00, vb0, c0);
        c1 = WMMA_BF16(a01, vb0, c1);
        c0 = WMMA_BF16(a10, vb1, c0);
        c1 = WMMA_BF16(a11, vb1, c1);
    }
}

// ---- Weight prep: bf16 conversion + both orientations --------------------
__global__ void prep_weights(const float* __restrict__ w1, const float* __restrict__ wsw,
                             const float* __restrict__ w2, const float* __restrict__ w3,
                             const float* __restrict__ w4, __bf16* __restrict__ wb) {
    const int idx = blockIdx.x * 256 + threadIdx.x;   // grid covers 262144
    if (idx < 262144) {
        const int k = idx >> 8, n = idx & 255;        // idx = k*256+n
        const float v1 = w1[idx], vs = wsw[idx];
        wb[(size_t)n * 1024 + k]          = (__bf16)v1;   // W1f (col-major)
        wb[262144 + idx]                  = (__bf16)v1;   // W1b (= col-major of W1^T)
        wb[524288 + (size_t)n * 1024 + k] = (__bf16)vs;   // Wsf
        wb[786432 + idx]                  = (__bf16)vs;   // Wsb
    }
    if (idx < 65536) {
        const int k = idx >> 8, n = idx & 255;
        const float v2 = w2[idx], v3 = w3[idx], v4 = w4[idx];
        wb[1048576 + n * 256 + k] = (__bf16)v2;
        wb[1114112 + idx]         = (__bf16)v2;
        wb[1179648 + n * 256 + k] = (__bf16)v3;
        wb[1245184 + idx]         = (__bf16)v3;
        wb[1310720 + n * 256 + k] = (__bf16)v4;
        wb[1376256 + idx]         = (__bf16)v4;
    }
}

// ---- Fully fused 3-step leapfrog, 32 rows per workgroup, 16 waves --------
__global__ __launch_bounds__(512)
void leapfrog_kernel(
    const float* __restrict__ x,
    const float* __restrict__ b1,  const float* __restrict__ g1v, const float* __restrict__ be1,
    const float* __restrict__ mm1, const float* __restrict__ mv1,
    const float* __restrict__ b2,  const float* __restrict__ bsv,
    const float* __restrict__ b3,  const float* __restrict__ g2v, const float* __restrict__ be2,
    const float* __restrict__ mm2v, const float* __restrict__ mv2,
    const float* __restrict__ b4,  const float* __restrict__ wf,
    const __bf16* __restrict__ W1f, const __bf16* __restrict__ W1b,
    const __bf16* __restrict__ Wsf, const __bf16* __restrict__ Wsb,
    const __bf16* __restrict__ W2f, const __bf16* __restrict__ W2b,
    const __bf16* __restrict__ W3f, const __bf16* __restrict__ W3b,
    const __bf16* __restrict__ W4f, const __bf16* __restrict__ W4b,
    float* __restrict__ out)
{
    extern __shared__ char smem[];
    float*  s_qp    = (float*)smem;                  // 32x1024 f32: [0,512)=q [512,1024)=p
    __bf16* s_state = (__bf16*)(smem + 131072);      // 32x1024 bf16
    __bf16* s_h1    = (__bf16*)(smem + 196608);      // 32x256 bf16 each below
    __bf16* s_h1bn  = (__bf16*)(smem + 212992);
    __bf16* s_h2    = (__bf16*)(smem + 229376);
    __bf16* s_out1  = (__bf16*)(smem + 245760);
    __bf16* s_h1p   = (__bf16*)(smem + 262144);
    __bf16* s_h1bn2 = (__bf16*)(smem + 278528);
    __bf16* s_g1    = (__bf16*)(smem + 294912);      // total 311296 B
    const int NTHR = 512;

    const int tid  = threadIdx.x;
    const int lane = tid & 31;
    const int wave = tid >> 5;
    const int row0 = blockIdx.x * 32;

    // q = x[:,3,:] ; p = x[:,3,:] - x[:,2,:]
    for (int i = tid; i < 32 * 512; i += NTHR) {
        const int m = i >> 9, d = i & 511;
        const float xl = x[(size_t)(row0 + m) * 2048 + 1536 + d];
        const float xm = x[(size_t)(row0 + m) * 2048 + 1024 + d];
        s_qp[m * 1024 + d]       = xl;
        s_qp[m * 1024 + 512 + d] = xl - xm;
    }
    __syncthreads();

    for (int it = 0; it < 6; ++it) {
        const bool firstHalf = ((it & 1) == 0);

        for (int i = tid; i < 32 * 1024; i += NTHR)
            s_state[i] = (__bf16)s_qp[i];
        __syncthreads();

        const int bc  = wave * 16;            // each wave owns one 16-col tile (N=256 phases)
        const int col = bc + (lane & 15);
        const int r0  = (lane >> 4) << 3;     // rows r0..r0+7 (m-tile0) / +16 (m-tile1)

        float st0[8], st1[8];                 // shortcut, kept in regs across one barrier

        // ---- fwd res1: h1 = tanh(state@W1+b1); h1bn; shortcut = state@Ws+bs
        {
            v8f z0 = {}, s0 = {}, z1 = {}, s1 = {};
            mm2x2<1024>(s_state, W1f, Wsf, bc, lane, z0, s0, z1, s1);
            const float bb = b1[col], sc = g1v[col] * rsq_fast(mv1[col] + BN_EPS_F);
            const float mu = mm1[col], bt = be1[col], bsh = bsv[col];
#pragma unroll
            for (int j = 0; j < 8; ++j) {
                const int ra = (r0 + j) * 256 + col, rb = (16 + r0 + j) * 256 + col;
                float h = tanh_fast(z0[j] + bb);
                s_h1[ra] = (__bf16)h;  s_h1bn[ra] = (__bf16)((h - mu) * sc + bt);
                st0[j] = s0[j] + bsh;
                h = tanh_fast(z1[j] + bb);
                s_h1[rb] = (__bf16)h;  s_h1bn[rb] = (__bf16)((h - mu) * sc + bt);
                st1[j] = s1[j] + bsh;
            }
        }
        __syncthreads();

        // ---- h2 = tanh(h1bn@W2+b2); out1 = h2 + shortcut
        {
            v8f z0 = {}, z1 = {};
            mm2<256>(s_h1bn, W2f, bc, lane, z0, z1);
            const float bb = b2[col];
#pragma unroll
            for (int j = 0; j < 8; ++j) {
                const int ra = (r0 + j) * 256 + col, rb = (16 + r0 + j) * 256 + col;
                float h = tanh_fast(z0[j] + bb);
                s_h2[ra] = (__bf16)h;  s_out1[ra] = (__bf16)(h + st0[j]);
                h = tanh_fast(z1[j] + bb);
                s_h2[rb] = (__bf16)h;  s_out1[rb] = (__bf16)(h + st1[j]);
            }
        }
        __syncthreads();

        // ---- res2 fwd: h1' = tanh(out1@W3+b3); h1bn'
        {
            v8f z0 = {}, z1 = {};
            mm2<256>(s_out1, W3f, bc, lane, z0, z1);
            const float bb = b3[col], sc = g2v[col] * rsq_fast(mv2[col] + BN_EPS_F);
            const float mu = mm2v[col], bt = be2[col];
#pragma unroll
            for (int j = 0; j < 8; ++j) {
                const int ra = (r0 + j) * 256 + col, rb = (16 + r0 + j) * 256 + col;
                float h = tanh_fast(z0[j] + bb);
                s_h1p[ra] = (__bf16)h;  s_h1bn2[ra] = (__bf16)((h - mu) * sc + bt);
                h = tanh_fast(z1[j] + bb);
                s_h1p[rb] = (__bf16)h;  s_h1bn2[rb] = (__bf16)((h - mu) * sc + bt);
            }
        }
        __syncthreads();

        // ---- h2' fused with bwd seed: g_z2' = wf * (1 - h2'^2)
        {
            v8f z0 = {}, z1 = {};
            mm2<256>(s_h1bn2, W4f, bc, lane, z0, z1);
            const float bb = b4[col], wfv = wf[col];
#pragma unroll
            for (int j = 0; j < 8; ++j) {
                const int ra = (r0 + j) * 256 + col, rb = (16 + r0 + j) * 256 + col;
                float h = tanh_fast(z0[j] + bb);
                s_g1[ra] = (__bf16)(wfv * (1.f - h * h));
                h = tanh_fast(z1[j] + bb);
                s_g1[rb] = (__bf16)(wfv * (1.f - h * h));
            }
        }
        __syncthreads();

        // ---- g_z1' = (g_z2' @ W4^T) * s2 * (1 - h1'^2)  -> reuse s_h1bn2
        {
            v8f g0 = {}, g1 = {};
            mm2<256>(s_g1, W4b, bc, lane, g0, g1);
            const float sc = g2v[col] * rsq_fast(mv2[col] + BN_EPS_F);
#pragma unroll
            for (int j = 0; j < 8; ++j) {
                const int ra = (r0 + j) * 256 + col, rb = (16 + r0 + j) * 256 + col;
                float h = (float)s_h1p[ra];
                s_h1bn2[ra] = (__bf16)(g0[j] * sc * (1.f - h * h));
                h = (float)s_h1p[rb];
                s_h1bn2[rb] = (__bf16)(g1[j] * sc * (1.f - h * h));
            }
        }
        __syncthreads();

        // ---- g_out1 = g_z1' @ W3^T + wf -> s_out1 ; g_z2 = g_out1*(1-h2^2) -> s_h1p
        {
            v8f g0 = {}, g1 = {};
            mm2<256>(s_h1bn2, W3b, bc, lane, g0, g1);
            const float wfv = wf[col];
#pragma unroll
            for (int j = 0; j < 8; ++j) {
                const int ra = (r0 + j) * 256 + col, rb = (16 + r0 + j) * 256 + col;
                float go = g0[j] + wfv;
                s_out1[ra] = (__bf16)go;
                float h = (float)s_h2[ra];
                s_h1p[ra] = (__bf16)(go * (1.f - h * h));
                go = g1[j] + wfv;
                s_out1[rb] = (__bf16)go;
                h = (float)s_h2[rb];
                s_h1p[rb] = (__bf16)(go * (1.f - h * h));
            }
        }
        __syncthreads();

        // ---- g_z1 = (g_z2 @ W2^T) * s1 * (1 - h1^2) -> s_h1bn
        {
            v8f g0 = {}, g1 = {};
            mm2<256>(s_h1p, W2b, bc, lane, g0, g1);
            const float sc = g1v[col] * rsq_fast(mv1[col] + BN_EPS_F);
#pragma unroll
            for (int j = 0; j < 8; ++j) {
                const int ra = (r0 + j) * 256 + col, rb = (16 + r0 + j) * 256 + col;
                float h = (float)s_h1[ra];
                s_h1bn[ra] = (__bf16)(g0[j] * sc * (1.f - h * h));
                h = (float)s_h1[rb];
                s_h1bn[rb] = (__bf16)(g1[j] * sc * (1.f - h * h));
            }
        }
        __syncthreads();

        // ---- g_state = g_z1 @ W1^T + g_out1 @ Ws^T, fused symplectic update.
        // Second half-step needs only dH[:, :512] -> half the columns.
        const int nt = firstHalf ? 4 : 2;
        const int cb = firstHalf ? wave * 64 : wave * 32;
        for (int t = 0; t < nt; ++t) {
            const int bcf = cb + t * 16;
            v8f g0 = {}, g1 = {};
            mmfin<256>(s_h1bn, s_out1, W1b, Wsb, bcf, lane, g0, g1);
            const int cf = bcf + (lane & 15);
            if (cf < 512) {                       // dH/dq -> p -= 0.5*dt*dH
#pragma unroll
                for (int j = 0; j < 8; ++j) {
                    s_qp[(r0 + j) * 1024 + 512 + cf]      -= HALF_DT * g0[j];
                    s_qp[(16 + r0 + j) * 1024 + 512 + cf] -= HALF_DT * g1[j];
                }
            } else if (firstHalf) {               // dH/dp -> q += dt*dH
#pragma unroll
                for (int j = 0; j < 8; ++j) {
                    s_qp[(r0 + j) * 1024 + cf - 512]      += DT_F * g0[j];
                    s_qp[(16 + r0 + j) * 1024 + cf - 512] += DT_F * g1[j];
                }
            }
        }
        __syncthreads();
    }

    for (int i = tid; i < 32 * 1024; i += NTHR)
        out[(size_t)row0 * 1024 + i] = s_qp[i];
}

extern "C" void kernel_launch(void* const* d_in, const int* in_sizes, int n_in,
                              void* d_out, int out_size, void* d_ws, size_t ws_size,
                              hipStream_t stream) {
    const float* x   = (const float*)d_in[0];
    const float* w1  = (const float*)d_in[1];
    const float* b1  = (const float*)d_in[2];
    const float* g1  = (const float*)d_in[3];
    const float* be1 = (const float*)d_in[4];
    const float* mm1 = (const float*)d_in[5];
    const float* mv1 = (const float*)d_in[6];
    const float* w2  = (const float*)d_in[7];
    const float* b2  = (const float*)d_in[8];
    const float* wsw = (const float*)d_in[9];
    const float* bs  = (const float*)d_in[10];
    const float* w3  = (const float*)d_in[11];
    const float* b3  = (const float*)d_in[12];
    const float* g2  = (const float*)d_in[13];
    const float* be2 = (const float*)d_in[14];
    const float* mm2 = (const float*)d_in[15];
    const float* mv2 = (const float*)d_in[16];
    const float* w4  = (const float*)d_in[17];
    const float* b4  = (const float*)d_in[18];
    const float* wf  = (const float*)d_in[19];
    // d_in[20] = bf: constant offset, does not affect gradients.

    __bf16* wb = (__bf16*)d_ws;   // needs 1,441,792 halfs = 2.75 MB
    prep_weights<<<1024, 256, 0, stream>>>(w1, wsw, w2, w3, w4, wb);

    const __bf16* W1f = wb;
    const __bf16* W1b = wb + 262144;
    const __bf16* Wsf = wb + 524288;
    const __bf16* Wsb = wb + 786432;
    const __bf16* W2f = wb + 1048576;
    const __bf16* W2b = wb + 1114112;
    const __bf16* W3f = wb + 1179648;
    const __bf16* W3b = wb + 1245184;
    const __bf16* W4f = wb + 1310720;
    const __bf16* W4b = wb + 1376256;

    leapfrog_kernel<<<1024, 512, 311296, stream>>>(
        x, b1, g1, be1, mm1, mv1, b2, bs,
        b3, g2, be2, mm2, mv2, b4, wf,
        W1f, W1b, Wsf, Wsb, W2f, W2b, W3f, W3b, W4f, W4b,
        (float*)d_out);
}
